// FreeYOLO_18949395710349
// MI455X (gfx1250) — matrix operand order repaired and checked by reading, not compiled, not run on hardware
//
#include <hip/hip_runtime.h>
#include <hip/hip_bf16.h>
#include <math.h>

typedef __attribute__((ext_vector_type(16))) _Float16 v16h;
typedef __attribute__((ext_vector_type(8)))  float    v8f;

#define IMG_F   2048.0f
#define NCLS    80
#define REGC    68
#define TOPKN   1000
#define NTOT    3000
#define CAP     4096
#define HBINS   65536
#define KCONF   0.05f
#define NMS_T   0.6f
#define NWORDS  96

// workspace byte offsets (total ~2.1 MB)
#define OFF_HIST 0u          // 3*65536*4 = 786432
#define OFF_CNT  786432u     // 3 u32 (+pad)
#define OFF_CUT  786496u     // 3 u32 (+pad)
#define OFF_CAND 786560u     // 3*CAP*8 = 98304
#define OFF_GBOX 884864u     // 3000*4 f32
#define OFF_GSC  932864u     // 3000 f32
#define OFF_GLB  944864u     // 3000 f32
#define OFF_GVD  956864u     // 3000 f32
#define OFF_SOB  968864u     // sorted offset boxes 3000*4 f32 (16B aligned)
#define OFF_SVD  1016864u    // sorted valid 3000 f32
#define OFF_SUP  1028864u    // suppression matrix 3000*96 u32

__device__ __forceinline__ unsigned fkey(float f) {
  unsigned u = __float_as_uint(f);
  return u ^ (unsigned)(((int)u >> 31) | (int)0x80000000);
}
__device__ __forceinline__ float fkeyinv(unsigned k) {
  unsigned u = k ^ ((k >> 31) ? 0x80000000u : 0xFFFFFFFFu);
  return __uint_as_float(u);
}
__device__ __forceinline__ float clip01(float v) {
  return fminf(fmaxf(v, 0.0f), 1.0f);
}

// In-LDS bitonic sort, descending, n power of two.
__device__ void bitonic_desc(unsigned long long* s, int n, int tid, int nthreads) {
  for (int k = 2; k <= n; k <<= 1) {
    for (int j = k >> 1; j > 0; j >>= 1) {
      for (int i = tid; i < n; i += nthreads) {
        int ixj = i ^ j;
        if (ixj > i) {
          unsigned long long a = s[i], b = s[ixj];
          bool up = ((i & k) == 0);
          bool sw = up ? (a < b) : (a > b);   // overall descending
          if (sw) { s[i] = b; s[ixj] = a; }
        }
      }
      __syncthreads();
    }
  }
}

// ---------------- kernels ----------------

__global__ void k_zero(char* ws) {
  unsigned* p = (unsigned*)ws;
  int n = (int)(OFF_CAND / 4);
  for (int i = blockIdx.x * blockDim.x + threadIdx.x; i < n; i += gridDim.x * blockDim.x)
    p[i] = 0u;
}

__global__ void k_hist(const float* __restrict__ cls, int n, unsigned* __restrict__ hist) {
  int i = blockIdx.x * blockDim.x + threadIdx.x;
  int st = gridDim.x * blockDim.x;
  for (; i < n; i += st) {
    __builtin_prefetch(cls + i + st, 0, 1);          // global_prefetch_b8
    unsigned key = fkey(cls[i]);
    atomicAdd(&hist[key >> 16], 1u);
  }
}

__global__ void __launch_bounds__(256) k_cutoff(char* ws) {
  int lev = blockIdx.x;
  const unsigned* hist = (const unsigned*)(ws + OFF_HIST) + (size_t)lev * HBINS;
  unsigned* cut = (unsigned*)(ws + OFF_CUT);
  __shared__ unsigned cs[256];
  int tid = threadIdx.x;
  unsigned sum = 0;
  for (int b = 0; b < 256; ++b) sum += hist[tid * 256 + b];
  cs[tid] = sum;
  __syncthreads();
  if (tid == 0) {
    unsigned acc = 0; unsigned cutbin = 0;
    for (int c = 255; c >= 0; --c) {
      if (acc + cs[c] >= TOPKN) {
        for (int b = c * 256 + 255; b >= c * 256; --b) {
          acc += hist[b];
          if (acc >= TOPKN) { cutbin = (unsigned)b; break; }
        }
        break;
      }
      acc += cs[c];
    }
    cut[lev] = cutbin;
  }
}

__global__ void k_compact(const float* __restrict__ cls, int n, char* ws, int lev) {
  unsigned cutbin = ((const unsigned*)(ws + OFF_CUT))[lev];
  unsigned* cnt = (unsigned*)(ws + OFF_CNT);
  unsigned long long* cand = (unsigned long long*)(ws + OFF_CAND) + (size_t)lev * CAP;
  int i = blockIdx.x * blockDim.x + threadIdx.x;
  int st = gridDim.x * blockDim.x;
  for (; i < n; i += st) {
    __builtin_prefetch(cls + i + st, 0, 1);
    unsigned key = fkey(cls[i]);
    if ((key >> 16) >= cutbin) {
      unsigned pos = atomicAdd(&cnt[lev], 1u);
      if (pos < CAP)
        cand[pos] = ((unsigned long long)key << 32) | (unsigned)(0xFFFFFFFFu - (unsigned)i);
    }
  }
}

// One block per level: sort candidates, pick top-1000, DFL decode via WMMA.
__global__ void __launch_bounds__(1024) k_finalize(char* ws,
                                                   const float* __restrict__ reg0,
                                                   const float* __restrict__ reg1,
                                                   const float* __restrict__ reg2) {
  int lev = blockIdx.x;
  const float* reg = (lev == 0) ? reg0 : ((lev == 1) ? reg1 : reg2);
  int strideLv = 8 << lev;       // 8,16,32
  int fdim = 256 >> lev;         // 256,128,64
  unsigned* cnt = (unsigned*)(ws + OFF_CNT);
  const unsigned long long* cand = (const unsigned long long*)(ws + OFF_CAND) + (size_t)lev * CAP;
  float* gbox = (float*)(ws + OFF_GBOX);
  float* gsc  = (float*)(ws + OFF_GSC);
  float* glb  = (float*)(ws + OFF_GLB);
  float* gvd  = (float*)(ws + OFF_GVD);

  __shared__ unsigned long long s[CAP];    // 32 KB
  int tid = threadIdx.x;
  unsigned n = cnt[lev]; if (n > CAP) n = CAP;
  for (int t = tid; t < CAP; t += 1024) s[t] = (t < (int)n) ? cand[t] : 0ull;
  __syncthreads();
  bitonic_desc(s, CAP, tid, 1024);

  // Phase A: scores / labels / valid for top-1000 (sorted desc, index tie-break asc).
  if (tid < TOPKN) {
    unsigned long long kk = s[tid];
    unsigned key = (unsigned)(kk >> 32);
    unsigned idx = 0xFFFFFFFFu - (unsigned)(kk & 0xFFFFFFFFu);
    float logit = fkeyinv(key);
    float sc = 1.0f / (1.0f + __expf(-logit));
    int g = lev * TOPKN + tid;
    gsc[g] = sc;
    glb[g] = (float)(idx % NCLS);
    gvd[g] = (sc > KCONF) ? 1.0f : 0.0f;
  }

  // Phase B: DFL projection via v_wmma_f32_16x16x32_f16.
  // 16 rows per wave = 4 candidates x 4 sides; K = 17 bins padded to 32.
  // A (16x32 f16): lane<16 holds row m=lane K={0..7,16..23}; lane>=16 row m=lane-16 K={8..15,24..31}.
  // B (32x16 f16): proj[k]=k (k<=16) broadcast over all 16 columns.
  int wave = tid >> 5, lane = tid & 31;
  bool lo = (lane < 16);
  for (int grp = wave; grp < TOPKN / 4; grp += 32) {
    int m = lane & 15;
    int c = grp * 4 + (m >> 2);
    int side = m & 3;
    unsigned long long kk = s[c];
    unsigned idx = 0xFFFFFFFFu - (unsigned)(kk & 0xFFFFFFFFu);
    int anchor = (int)(idx / NCLS);
    const float* rp = reg + (size_t)anchor * REGC + side * 17;

    float p[17];
    float mx = rp[0];
    #pragma unroll
    for (int b = 1; b < 17; ++b) mx = fmaxf(mx, rp[b]);
    float sum = 0.0f;
    #pragma unroll
    for (int b = 0; b < 17; ++b) { p[b] = __expf(rp[b] - mx); sum += p[b]; }
    float inv = 1.0f / sum;

    v16h a, bf;
    #pragma unroll
    for (int e = 0; e < 16; ++e) {
      int Klo = (e < 8) ? e : (e + 8);
      int Khi = (e < 8) ? (e + 8) : (e + 16);
      int K = lo ? Klo : Khi;
      float av = (K <= 16) ? (p[K <= 16 ? K : 0] * inv) : 0.0f;
      a[e] = (_Float16)av;
      int Kb = K;
      bf[e] = (_Float16)((Kb <= 16) ? (float)Kb : 0.0f);
    }
    v8f cacc = {};
    v8f d = __builtin_amdgcn_wmma_f32_16x16x32_f16(false, a, false, bf,
                                                   (short)0, cacc, false, false);
    // D layout: lane holds rows r+(lane>=16?8:0) in d[r], identical across N -> take N==0 lanes.
    if ((lane & 15) == 0) {
      int baseRow = lo ? 0 : 8;
      #pragma unroll
      for (int cc = 0; cc < 2; ++cc) {
        int candIdx = grp * 4 + baseRow / 4 + cc;
        unsigned long long k2 = s[candIdx];
        unsigned idx2 = 0xFFFFFFFFu - (unsigned)(k2 & 0xFFFFFFFFu);
        int anc = (int)(idx2 / NCLS);
        float ax = ((float)(anc % fdim) + 0.5f) * (float)strideLv;
        float ay = ((float)(anc / fdim) + 0.5f) * (float)strideLv;
        float d0 = d[cc * 4 + 0], d1 = d[cc * 4 + 1];
        float d2 = d[cc * 4 + 2], d3 = d[cc * 4 + 3];
        int g = lev * TOPKN + candIdx;
        gbox[g * 4 + 0] = ax - d0 * (float)strideLv;
        gbox[g * 4 + 1] = ay - d1 * (float)strideLv;
        gbox[g * 4 + 2] = ax + d2 * (float)strideLv;
        gbox[g * 4 + 3] = ay + d3 * (float)strideLv;
      }
    }
  }
}

// Global sort of 3000 candidates (valid first, score desc, index asc), emit outputs + NMS inputs.
__global__ void __launch_bounds__(1024) k_gsort(char* ws, float* dout) {
  const float* gbox = (const float*)(ws + OFF_GBOX);
  const float* gsc  = (const float*)(ws + OFF_GSC);
  const float* glb  = (const float*)(ws + OFF_GLB);
  const float* gvd  = (const float*)(ws + OFF_GVD);
  float* sob = (float*)(ws + OFF_SOB);
  float* svd = (float*)(ws + OFF_SVD);
  __shared__ unsigned long long s[4096];   // 32 KB
  int tid = threadIdx.x;
  for (int t = tid; t < 4096; t += 1024) {
    unsigned long long key = 0ull;
    if (t < NTOT) {
      unsigned k32 = (gvd[t] > 0.5f) ? fkey(gsc[t]) : 0u;
      key = ((unsigned long long)k32 << 32) | (unsigned)(4095 - t);
    }
    s[t] = key;
  }
  __syncthreads();
  bitonic_desc(s, 4096, tid, 1024);

  float* obb = dout;            // [3000,4]
  float* osc = dout + 12000;    // [3000]
  float* olb = dout + 15000;    // [3000]
  for (int t = tid; t < NTOT; t += 1024) {
    unsigned long long kk = s[t];
    int i = 4095 - (int)(unsigned)(kk & 0xFFFFFFFFu);
    float b0 = gbox[i * 4 + 0], b1 = gbox[i * 4 + 1];
    float b2 = gbox[i * 4 + 2], b3 = gbox[i * 4 + 3];
    obb[t * 4 + 0] = clip01(b0 * (1.0f / IMG_F));
    obb[t * 4 + 1] = clip01(b1 * (1.0f / IMG_F));
    obb[t * 4 + 2] = clip01(b2 * (1.0f / IMG_F));
    obb[t * 4 + 3] = clip01(b3 * (1.0f / IMG_F));
    osc[t] = gsc[i];
    float lb = glb[i];
    olb[t] = lb;
    float off = lb * (4.0f * IMG_F);      // class-offset trick
    sob[t * 4 + 0] = b0 + off; sob[t * 4 + 1] = b1 + off;
    sob[t * 4 + 2] = b2 + off; sob[t * 4 + 3] = b3 + off;
    svd[t] = gvd[i];
  }
}

// One block per row i: suppression bitmask over j. Boxes staged via async-to-LDS.
__global__ void __launch_bounds__(256) k_supmat(char* ws) {
  int i = blockIdx.x;
  const float4* sobg = (const float4*)(ws + OFF_SOB);
  unsigned* sup = (unsigned*)(ws + OFF_SUP) + (size_t)i * NWORDS;
  __shared__ float4 sb[NTOT];              // 48 KB
  __shared__ unsigned words[NWORDS];
  int tid = threadIdx.x;
  if (tid < NWORDS) words[tid] = 0u;
  // CDNA5 async copy: global -> LDS, tracked by ASYNCcnt.
  for (int t = tid; t < NTOT; t += 256) {
    unsigned ldsoff = (unsigned)(unsigned long long)(size_t)(&sb[t]); // low 32 bits = LDS offset
    unsigned long long ga = (unsigned long long)(size_t)(&sobg[t]);
    asm volatile("global_load_async_to_lds_b128 %0, %1, off"
                 :: "v"(ldsoff), "v"(ga) : "memory");
  }
  asm volatile("s_wait_asynccnt 0" ::: "memory");
  __syncthreads();

  float4 bi = sb[i];
  float areai = fmaxf(bi.z - bi.x, 0.0f) * fmaxf(bi.w - bi.y, 0.0f);
  for (int j = tid; j < NTOT; j += 256) {
    if (j > i) {
      float4 bj = sb[j];
      float ltx = fmaxf(bi.x, bj.x), lty = fmaxf(bi.y, bj.y);
      float rbx = fminf(bi.z, bj.z), rby = fminf(bi.w, bj.w);
      float inter = fmaxf(rbx - ltx, 0.0f) * fmaxf(rby - lty, 0.0f);
      float areaj = fmaxf(bj.z - bj.x, 0.0f) * fmaxf(bj.w - bj.y, 0.0f);
      float uni = fmaxf(areai + areaj - inter, 1e-9f);
      if (inter / uni > NMS_T) atomicOr(&words[j >> 5], 1u << (j & 31));
    }
  }
  __syncthreads();
  if (tid < NWORDS) sup[tid] = words[tid];
}

// Single wave32: sequential bitmask scan (lockstep, no barriers), writes keep flags.
__global__ void __launch_bounds__(32) k_scan(char* ws, float* out_keep) {
  const unsigned* sup = (const unsigned*)(ws + OFF_SUP);
  const float* svd = (const float*)(ws + OFF_SVD);
  __shared__ unsigned keepw_s[NWORDS];
  volatile unsigned* keepw = keepw_s;
  int l = threadIdx.x;
  for (int w = l; w < NWORDS; w += 32) {
    unsigned m = 0;
    for (int b = 0; b < 32; ++b) {
      int idx = w * 32 + b;
      if (idx < NTOT && svd[idx] > 0.5f) m |= (1u << b);
    }
    keepw[w] = m;
  }
  asm volatile("s_wait_dscnt 0" ::: "memory");
  for (int i = 0; i < NTOT; ++i) {
    unsigned kw = keepw[i >> 5];
    if ((kw >> (i & 31)) & 1u) {
      const unsigned* row = sup + (size_t)i * NWORDS;
      __builtin_prefetch(sup + (size_t)(i + 1) * NWORDS, 0, 1);
      for (int w = l; w < NWORDS; w += 32) keepw[w] &= ~row[w];
    }
  }
  for (int idx = l; idx < NTOT; idx += 32)
    out_keep[idx] = ((keepw[idx >> 5] >> (idx & 31)) & 1u) ? 1.0f : 0.0f;
}

extern "C" void kernel_launch(void* const* d_in, const int* in_sizes, int n_in,
                              void* d_out, int out_size, void* d_ws, size_t ws_size,
                              hipStream_t stream) {
  if (n_in < 6) return;
  const float* cls[3];
  const float* reg[3];
  // setup_inputs() dict order interleaves cls_i/reg_i; detect via sizes (reg0 = 65536*68).
  if (in_sizes[1] == 65536 * 68) {
    cls[0] = (const float*)d_in[0]; reg[0] = (const float*)d_in[1];
    cls[1] = (const float*)d_in[2]; reg[1] = (const float*)d_in[3];
    cls[2] = (const float*)d_in[4]; reg[2] = (const float*)d_in[5];
  } else {
    cls[0] = (const float*)d_in[0]; cls[1] = (const float*)d_in[1]; cls[2] = (const float*)d_in[2];
    reg[0] = (const float*)d_in[3]; reg[1] = (const float*)d_in[4]; reg[2] = (const float*)d_in[5];
  }
  char* ws = (char*)d_ws;
  float* dout = (float*)d_out;

  const int MC[3] = {65536 * NCLS, 16384 * NCLS, 4096 * NCLS};

  k_zero<<<512, 256, 0, stream>>>(ws);
  for (int l = 0; l < 3; ++l) {
    unsigned* hist = (unsigned*)(ws + OFF_HIST) + (size_t)l * HBINS;
    k_hist<<<2048, 256, 0, stream>>>(cls[l], MC[l], hist);
  }
  k_cutoff<<<3, 256, 0, stream>>>(ws);
  for (int l = 0; l < 3; ++l)
    k_compact<<<2048, 256, 0, stream>>>(cls[l], MC[l], ws, l);
  k_finalize<<<3, 1024, 0, stream>>>(ws, reg[0], reg[1], reg[2]);
  k_gsort<<<1, 1024, 0, stream>>>(ws, dout);
  k_supmat<<<NTOT, 256, 0, stream>>>(ws);
  k_scan<<<1, 32, 0, stream>>>(ws, dout + 18000);
}
